// SpikingMultiHeadAttention_48550310314471
// MI455X (gfx1250) — matrix operand order repaired and checked by reading, not compiled, not run on hardware
//
#include <hip/hip_runtime.h>
#include <hip/hip_bf16.h>
#include <stdint.h>

// ---------------------------------------------------------------------------
// Spiking MHA for MI455X (gfx1250, wave32).  B=2,S=2048,D=1024,H=16,hd=64.
//   out  = ctx @ Wo^T + bo           -> d_out[0 .. 4194304)           fp32
//   attn = heaviside(q@k^T * scale)  -> d_out[4194304 .. +134217728)  fp32
// Roofline: mandatory 537MB attn write (~23us @23.3TB/s) dominates; all GEMMs
// via v_wmma_f32_16x16x32_f16.  attn stores are packed b128 + non-temporal so
// the write stream does not evict the 40MB f16 working set from the 192MB L2.
// Tile staging uses gfx1250 global_load_async_to_lds_b128 (ASYNCcnt).
// ---------------------------------------------------------------------------

typedef __attribute__((ext_vector_type(16))) _Float16 v16h;
typedef __attribute__((ext_vector_type(8)))  float    v8f;
typedef __attribute__((ext_vector_type(4)))  float    f32x4;

union Frag16 { v16h h; uint32_t u[8]; };
union Pack4  { _Float16 h[4]; uint2 v; };

__device__ __forceinline__ v8f wmma_f16(v16h a, v16h b, v8f c) {
  return __builtin_amdgcn_wmma_f32_16x16x32_f16(false, a, false, b,
                                                (short)0, c, false, false);
}

// Async global->LDS copy, 16B per lane (tracked by ASYNCcnt).
__device__ __forceinline__ void async_load_b128(uint32_t lds_off,
                                                const void* gptr) {
  asm volatile("global_load_async_to_lds_b128 %0, %1, off"
               :: "v"(lds_off), "v"(gptr) : "memory");
}
__device__ __forceinline__ void wait_async0() {
  asm volatile("s_wait_asynccnt 0x0" ::: "memory");
}
// Low 32 bits of a generic pointer to __shared__ == LDS byte offset.
__device__ __forceinline__ uint32_t lds_off(const void* p) {
  return (uint32_t)(uintptr_t)p;
}

// A-matrix 16x32 f16 fragment from a row-major LDS tile (ld in halves).
// lanes 0-15: row lane&15, K={0..15}; lanes 16-31: K={16..31}.
__device__ __forceinline__ v16h lds_load_A(const _Float16* base, int rowBase,
                                           int ld, int kBase) {
  const int lane = threadIdx.x & 31;
  const int row  = rowBase + (lane & 15);
  const int hsel = (lane >> 4) & 1;
  const uint32_t* p = (const uint32_t*)base;
  Frag16 f;
  const int o = row * ld + kBase + hsel * 16;
#pragma unroll
  for (int j = 0; j < 8; ++j) f.u[j] = p[(o + 2 * j) >> 1];
  return f.h;
}

// B-matrix 32x16 f16 fragment; LDS tile row n holds output-column n along K.
__device__ __forceinline__ v16h lds_load_B(const _Float16* base, int colBase,
                                           int ld, int kBase) {
  const int lane = threadIdx.x & 31;
  const int col  = colBase + (lane & 15);
  const int kh   = (lane >> 4) & 1;
  const uint32_t* p = (const uint32_t*)base;
  Frag16 f;
  const int o = col * ld + kBase + kh * 16;
#pragma unroll
  for (int j = 0; j < 8; ++j) f.u[j] = p[(o + 2 * j) >> 1];
  return f.h;
}

// ---------------------------------------------------------------------------
// Kernel 0: one-shot f32->f16 conversion of the four weight matrices, so the
// GEMM kernels stream f16 tiles straight into LDS with async loads.
// ---------------------------------------------------------------------------
__global__ __launch_bounds__(256) void convert_w_kernel(
    const float* __restrict__ Wq, const float* __restrict__ Wk,
    const float* __restrict__ Wv, const float* __restrict__ Wo,
    _Float16* __restrict__ dq, _Float16* __restrict__ dk,
    _Float16* __restrict__ dv, _Float16* __restrict__ do_) {
  const int z = blockIdx.z;
  const float* src = (z == 0) ? Wq : (z == 1) ? Wk : (z == 2) ? Wv : Wo;
  _Float16*    dst = (z == 0) ? dq : (z == 1) ? dk : (z == 2) ? dv : do_;
  const size_t i = ((size_t)blockIdx.x * 256 + threadIdx.x) * 4;
  float4 v = *(const float4*)(src + i);
  Pack4 p;
  p.h[0] = (_Float16)v.x; p.h[1] = (_Float16)v.y;
  p.h[2] = (_Float16)v.z; p.h[3] = (_Float16)v.w;
  *(uint2*)(dst + i) = p.v;
}

// ---------------------------------------------------------------------------
// Kernel 1: fused q/k/v projection.  Y = X @ W^T + b  (M=4096,N=1024,K=1024)
// X: f32 (converted inline), W: f16 (async tiles). Output f16 [B,H,S,hd].
// ---------------------------------------------------------------------------
__global__ __launch_bounds__(128) void qkv_proj_kernel(
    const float* __restrict__ qin, const float* __restrict__ kin,
    const float* __restrict__ vin,
    const _Float16* __restrict__ Wqh, const _Float16* __restrict__ Wkh,
    const _Float16* __restrict__ Wvh,
    const float* __restrict__ bq, const float* __restrict__ bk,
    const float* __restrict__ bv,
    _Float16* __restrict__ qo, _Float16* __restrict__ ko,
    _Float16* __restrict__ vo) {
  __shared__ __align__(16) _Float16 Xt[64 * 32];
  __shared__ __align__(16) _Float16 Wt[64 * 32];
  const int t = threadIdx.x, wave = t >> 5;
  const int z = blockIdx.z;
  const float*    X    = (z == 0) ? qin : (z == 1) ? kin : vin;
  const _Float16* W    = (z == 0) ? Wqh : (z == 1) ? Wkh : Wvh;
  const float*    bias = (z == 0) ? bq  : (z == 1) ? bk  : bv;
  _Float16*       Y    = (z == 0) ? qo  : (z == 1) ? ko  : vo;
  const int n0 = blockIdx.x * 64, m0 = blockIdx.y * 64;
  const uint32_t wt_off = lds_off(Wt);

  v8f acc[4] = {};
  for (int kk = 0; kk < 1024; kk += 32) {
    __syncthreads();
    // W tile 64x32 f16: async, 256 x 16B chunks (rows strided 1024 halves)
#pragma unroll
    for (int j = 0; j < 2; ++j) {
      const int c = t + j * 128;
      const int r = c >> 2, sub = c & 3;
      async_load_b128(wt_off + c * 16,
                      (const char*)W + ((size_t)(n0 + r) * 1024 + kk) * 2 +
                          sub * 16);
    }
    // X tile 64x32 f32 -> f16 via VGPRs
#pragma unroll
    for (int j = 0; j < 4; ++j) {
      const int idx = j * 512 + t * 4;
      const int r = idx >> 5, c = idx & 31;
      float4 xv = *(const float4*)(X + (size_t)(m0 + r) * 1024 + kk + c);
      Pack4 px;
      px.h[0] = (_Float16)xv.x; px.h[1] = (_Float16)xv.y;
      px.h[2] = (_Float16)xv.z; px.h[3] = (_Float16)xv.w;
      *(uint2*)&Xt[idx] = px.v;
    }
    wait_async0();
    __syncthreads();
    v16h bf[4];
#pragma unroll
    for (int nb = 0; nb < 4; ++nb) bf[nb] = lds_load_B(Wt, nb * 16, 32, 0);
    v16h a = lds_load_A(Xt, wave * 16, 32, 0);
#pragma unroll
    for (int nb = 0; nb < 4; ++nb) acc[nb] = wmma_f16(a, bf[nb], acc[nb]);
  }
  const int lane = t & 31, col0 = lane & 15, rb = (lane >> 4) * 8;
#pragma unroll
  for (int nb = 0; nb < 4; ++nb) {
    const int n = n0 + nb * 16 + col0;
    const float bb = bias[n];
    const int h = n >> 6, d = n & 63;
#pragma unroll
    for (int i = 0; i < 8; ++i) {
      const int m = m0 + wave * 16 + rb + i;
      const int bidx = m >> 11, s = m & 2047;
      Y[((size_t)((bidx * 16 + h) * 2048 + s)) * 64 + d] =
          (_Float16)(acc[nb][i] + bb);
    }
  }
}

// ---------------------------------------------------------------------------
// Kernel 2: fused spike attention per (b,h,64 q-rows) workgroup.
// ---------------------------------------------------------------------------
__global__ __launch_bounds__(128) void spike_attn_kernel(
    const _Float16* __restrict__ qws, const _Float16* __restrict__ kws,
    const _Float16* __restrict__ vws, float* __restrict__ attn_out,
    _Float16* __restrict__ ctx_out) {
  __shared__ __align__(16) _Float16 Qt[64 * 64];     // [qrow][d]
  __shared__ __align__(16) _Float16 Kt[32 * 64];     // [kcol][d]
  __shared__ __align__(16) _Float16 Vt[64 * 32];     // transposed [d][kk]
  __shared__ __align__(16) _Float16 At[4 * 16 * 32]; // per-wave 0/1 tiles

  const int t = threadIdx.x, wave = t >> 5, lane = t & 31;
  const int q0 = blockIdx.x * 64;
  const int h = blockIdx.y, b = blockIdx.z;
  const size_t headBase = (size_t)(b * 16 + h) * 2048 * 64;
  const _Float16* Q = qws + headBase;
  const _Float16* K = kws + headBase;
  const _Float16* V = vws + headBase;
  float* A = attn_out + (size_t)(b * 16 + h) * 2048 * 2048;
  const uint32_t qt_off = lds_off(Qt), kt_off = lds_off(Kt);

  // Q block (8KB, contiguous for this head): async copy
#pragma unroll
  for (int j = 0; j < 4; ++j) {
    const int c = t + j * 128;
    async_load_b128(qt_off + c * 16,
                    (const char*)(Q + (size_t)q0 * 64) + c * 16);
  }
  wait_async0();
  __syncthreads();
  const v16h qa0 = lds_load_A(Qt, wave * 16, 64, 0);
  const v16h qa1 = lds_load_A(Qt, wave * 16, 64, 32);

  v8f ctx[4] = {};
  const int col0 = lane & 15, rb = (lane >> 4) * 8;
  const int prow = lane >> 1, pch = (lane & 1) * 16;  // packed attn-store map

  for (int kb = 0; kb < 64; ++kb) {
    const int k0 = kb * 32;
    __syncthreads();                       // protect Kt/Vt/At from overwrite
    // K block (4KB, contiguous): async copy
#pragma unroll
    for (int j = 0; j < 2; ++j) {
      const int c = t + j * 128;
      async_load_b128(kt_off + c * 16,
                      (const char*)(K + (size_t)k0 * 64) + c * 16);
    }
    // V block: load via VGPRs and transpose into Vt[d][kk]
#pragma unroll
    for (int j = 0; j < 4; ++j) {
      const int idx = j * 512 + t * 4;
      Pack4 pv; pv.v = *(const uint2*)&V[(size_t)k0 * 64 + idx];
      const int kkl = idx >> 6, d0 = idx & 63;
#pragma unroll
      for (int x = 0; x < 4; ++x) Vt[(d0 + x) * 32 + kkl] = pv.h[x];
    }
    wait_async0();
    __syncthreads();
    // scores (q@k^T) -> spike 0/1 into At (f16)
#pragma unroll
    for (int kt = 0; kt < 2; ++kt) {
      v16h kb0 = lds_load_B(Kt, kt * 16, 64, 0);
      v16h kb1 = lds_load_B(Kt, kt * 16, 64, 32);
      v8f s = {};
      s = wmma_f16(qa0, kb0, s);
      s = wmma_f16(qa1, kb1, s);
#pragma unroll
      for (int i = 0; i < 8; ++i) {        // scale>0: sign-only threshold
        At[wave * 512 + (rb + i) * 32 + kt * 16 + col0] =
            (_Float16)((s[i] > 0.0f) ? 1.0f : 0.0f);
      }
    }
    __syncthreads();                       // At visible (all waves in step)
    // packed non-temporal attn write: 4 x b128 per lane (one-way 537MB stream)
    {
      const uint2* ap = (const uint2*)&At[wave * 512 + prow * 32 + pch];
      float* arow = A + (size_t)(q0 + wave * 16 + prow) * 2048 + k0 + pch;
#pragma unroll
      for (int j = 0; j < 4; ++j) {
        Pack4 pk; pk.v = ap[j];
        f32x4 w = {(float)pk.h[0], (float)pk.h[1], (float)pk.h[2],
                   (float)pk.h[3]};
        __builtin_nontemporal_store(w, (f32x4*)(arow + 4 * j));
      }
    }
    // ctx += attn @ v
    const v16h af = lds_load_A(At + wave * 512, 0, 32, 0);
#pragma unroll
    for (int dt = 0; dt < 4; ++dt) {
      ctx[dt] = wmma_f16(af, lds_load_B(Vt, dt * 16, 32, 0), ctx[dt]);
    }
  }
  // ctx -> ws f16 [B,S,D]
#pragma unroll
  for (int dt = 0; dt < 4; ++dt) {
#pragma unroll
    for (int i = 0; i < 8; ++i) {
      const int qr = q0 + wave * 16 + rb + i;
      const int d = dt * 16 + col0;
      ctx_out[((size_t)(b * 2048 + qr)) * 1024 + h * 64 + d] =
          (_Float16)ctx[dt][i];
    }
  }
}

// ---------------------------------------------------------------------------
// Kernel 3: out = ctx @ Wo^T + bo (f16 tiles fully async, fp32 NT out)
// ---------------------------------------------------------------------------
__global__ __launch_bounds__(128) void out_proj_kernel(
    const _Float16* __restrict__ ctx, const _Float16* __restrict__ Woh,
    const float* __restrict__ bo, float* __restrict__ out) {
  __shared__ __align__(16) _Float16 Xt[64 * 32];
  __shared__ __align__(16) _Float16 Wt[64 * 32];
  const int t = threadIdx.x, wave = t >> 5;
  const int n0 = blockIdx.x * 64, m0 = blockIdx.y * 64;
  const uint32_t xt_off = lds_off(Xt), wt_off = lds_off(Wt);
  v8f acc[4] = {};
  for (int kk = 0; kk < 1024; kk += 32) {
    __syncthreads();
#pragma unroll
    for (int j = 0; j < 2; ++j) {
      const int c = t + j * 128;
      const int r = c >> 2, sub = c & 3;
      async_load_b128(xt_off + c * 16,
                      (const char*)ctx + ((size_t)(m0 + r) * 1024 + kk) * 2 +
                          sub * 16);
      async_load_b128(wt_off + c * 16,
                      (const char*)Woh + ((size_t)(n0 + r) * 1024 + kk) * 2 +
                          sub * 16);
    }
    wait_async0();
    __syncthreads();
    v16h bf[4];
#pragma unroll
    for (int nb = 0; nb < 4; ++nb) bf[nb] = lds_load_B(Wt, nb * 16, 32, 0);
    v16h a = lds_load_A(Xt, wave * 16, 32, 0);
#pragma unroll
    for (int nb = 0; nb < 4; ++nb) acc[nb] = wmma_f16(a, bf[nb], acc[nb]);
  }
  const int lane = t & 31, col0 = lane & 15, rb = (lane >> 4) * 8;
#pragma unroll
  for (int nb = 0; nb < 4; ++nb) {
    const int n = n0 + nb * 16 + col0;
    const float bb = bo[n];
#pragma unroll
    for (int i = 0; i < 8; ++i) {
      const int m = m0 + wave * 16 + rb + i;
      __builtin_nontemporal_store(acc[nb][i] + bb, out + (size_t)m * 1024 + n);
    }
  }
}

// ---------------------------------------------------------------------------
extern "C" void kernel_launch(void* const* d_in, const int* in_sizes, int n_in,
                              void* d_out, int out_size, void* d_ws,
                              size_t ws_size, hipStream_t stream) {
  const float* query = (const float*)d_in[0];
  const float* key   = (const float*)d_in[1];
  const float* value = (const float*)d_in[2];
  const float* Wq = (const float*)d_in[3];
  const float* bq = (const float*)d_in[4];
  const float* Wk = (const float*)d_in[5];
  const float* bk = (const float*)d_in[6];
  const float* Wv = (const float*)d_in[7];
  const float* bv = (const float*)d_in[8];
  const float* Wo = (const float*)d_in[9];
  const float* bo = (const float*)d_in[10];

  float* out  = (float*)d_out;                      // [2,2048,1024]
  float* attn = out + (size_t)2 * 2048 * 1024;      // [2,16,2048,2048]

  const size_t QKV = (size_t)4096 * 1024;           // halves per tensor
  const size_t WSZ = (size_t)1024 * 1024;           // halves per weight
  _Float16* qws = (_Float16*)d_ws;
  _Float16* kws = qws + QKV;
  _Float16* vws = kws + QKV;
  _Float16* cws = vws + QKV;
  _Float16* wqh = cws + QKV;
  _Float16* wkh = wqh + WSZ;
  _Float16* wvh = wkh + WSZ;
  _Float16* woh = wvh + WSZ;                        // total 40 MiB ws

  convert_w_kernel<<<dim3(1024, 1, 4), 256, 0, stream>>>(Wq, Wk, Wv, Wo, wqh,
                                                         wkh, wvh, woh);
  qkv_proj_kernel<<<dim3(16, 64, 3), 128, 0, stream>>>(
      query, key, value, wqh, wkh, wvh, bq, bk, bv, qws, kws, vws);
  spike_attn_kernel<<<dim3(32, 16, 2), 128, 0, stream>>>(qws, kws, vws, attn,
                                                         cws);
  out_proj_kernel<<<dim3(16, 64, 1), 128, 0, stream>>>(cws, Wo ? woh : woh, bo,
                                                       out);
}
